// Block_18717467476160
// MI455X (gfx1250) — compile-verified, matrix operand order checked by reading, not run on hardware
//
#include <hip/hip_runtime.h>
#include <cstdint>
#include <cstddef>

// ---------------------------------------------------------------- types
typedef __bf16 bf16;
typedef __attribute__((ext_vector_type(16))) __bf16 v16bf;
typedef __attribute__((ext_vector_type(8)))  float  v8f;

union FragU { v16bf v; uint4 q[2]; };

// ---------------------------------------------------------------- helpers
__device__ __forceinline__ v8f wmma_bf16(v16bf a, v16bf b, v8f c) {
  // D = A(16x32 bf16) x B(32x16 bf16) + C(16x16 f32)
  return __builtin_amdgcn_wmma_f32_16x16x32_bf16(false, a, false, b, (short)0, c,
                                                 false, false);
}

// A-fragment (and B-fragment via N-major transposed operand) loader.
// Per CDNA5 ISA 16-bit A-matrix 16x32 layout:
//   lanes 0-15 : row = lane,     elems 0..7 = K+0..7,  elems 8..15 = K+16..23
//   lanes 16-31: row = lane-16,  elems 0..7 = K+8..15, elems 8..15 = K+24..31
__device__ __forceinline__ v16bf ld_frag(const bf16* __restrict__ p, int ld,
                                         int row0, int k0) {
  const int lane = threadIdx.x & 31;
  const int r    = lane & 15;
  const int kb   = k0 + ((lane >> 4) << 3);
  const bf16* b  = p + (size_t)(row0 + r) * (size_t)ld + kb;
  FragU f;
  f.q[0] = *reinterpret_cast<const uint4*>(b);
  f.q[1] = *reinterpret_cast<const uint4*>(b + 16);
  return f.v;
}

__device__ __forceinline__ float gelu_exact(float x) {
  return 0.5f * x * (1.0f + erff(x * 0.7071067811865475f));
}

// block reduction (blockDim multiple of 32); sred must hold >= nwaves floats
__device__ __forceinline__ float blk_reduce(float v, float* sred, int nwaves) {
  const int lane = threadIdx.x & 31;
  const int wv   = threadIdx.x >> 5;
#pragma unroll
  for (int m = 16; m >= 1; m >>= 1) v += __shfl_xor(v, m);
  if (lane == 0) sred[wv] = v;
  __syncthreads();
  if (threadIdx.x == 0) {
    float s = 0.f;
    for (int i = 0; i < nwaves; ++i) s += sred[i];
    sred[0] = s;
  }
  __syncthreads();
  float r = sred[0];
  __syncthreads();
  return r;
}

// ---------------------------------------------------------------- weight prep
__global__ void k_wt_transpose(const float* __restrict__ W, bf16* __restrict__ Wt,
                               int K, int N) {
  // W is [K][N] row-major, Wt is [N][K] (N-major, contiguous K)
  size_t i = (size_t)blockIdx.x * 256 + threadIdx.x;
  if (i >= (size_t)K * N) return;
  int n = (int)(i / K);
  int k = (int)(i % K);
  Wt[i] = (bf16)W[(size_t)k * N + n];
}

__global__ void k_wt_convert(const float* __restrict__ W, bf16* __restrict__ Wt,
                             int count) {
  int i = blockIdx.x * 256 + threadIdx.x;
  if (i >= count) return;
  Wt[i] = (bf16)W[i];
}

__global__ void k_wt_conv2(const float* __restrict__ W, bf16* __restrict__ Wt) {
  // W [O=384][I=384][3][3]  ->  Wt [o][tap][i]  (tap = ky*3+kx)
  int i = blockIdx.x * 256 + threadIdx.x;
  if (i >= 384 * 3456) return;
  int o = i / 3456, rest = i % 3456;
  int tap = rest / 384, ci = rest % 384;
  Wt[i] = (bf16)W[(size_t)o * 3456 + (size_t)ci * 9 + tap];
}

// ---------------------------------------------------------------- LN1 + window partition
__global__ __launch_bounds__(256)
void k_ln1win(const float* __restrict__ x, const float* __restrict__ g,
              const float* __restrict__ b, bf16* __restrict__ out) {
  __shared__ float sred[8];
  const int tkn = blockIdx.x;          // 0..39199  (win*196 + tok)
  const int win = tkn / 196, tok = tkn % 196;
  const int bb = win / 25, wy = (win / 5) % 5, wx = win % 5;
  const int ty = tok / 14, tx = tok % 14;
  const int y = wy * 14 + ty, xx = wx * 14 + tx;
  bf16* orow = out + (size_t)tkn * 768;
  if (y >= 64 || xx >= 64) {           // zero-padded window token
    for (int c = threadIdx.x; c < 768; c += 256) orow[c] = (bf16)0.f;
    return;
  }
  const float* xr = x + (((size_t)bb * 64 + y) * 64 + xx) * 768;
  const int t = threadIdx.x;
  float v0 = xr[t], v1 = xr[t + 256], v2 = xr[t + 512];
  float mean = blk_reduce(v0 + v1 + v2, sred, 8) * (1.f / 768.f);
  float d0 = v0 - mean, d1 = v1 - mean, d2 = v2 - mean;
  float var = blk_reduce(d0 * d0 + d1 * d1 + d2 * d2, sred, 8) * (1.f / 768.f);
  float inv = rsqrtf(var + 1e-5f);
  orow[t]       = (bf16)(d0 * inv * g[t]       + b[t]);
  orow[t + 256] = (bf16)(d1 * inv * g[t + 256] + b[t + 256]);
  orow[t + 512] = (bf16)(d2 * inv * g[t + 512] + b[t + 512]);
}

// ---------------------------------------------------------------- generic WMMA GEMM
// C[M][N] = A(bf16 [M][K]) x Bt(bf16 [N][K], N-major) (+bias)(+gelu)(+resid)
// wave tile: 32x64 (8 WMMA / K-step, 6 fragment loads) ; block: 4x2 waves = 128x128
template <int BIAS, int GELU, int WF32, int WBF16, int RESID>
__global__ __launch_bounds__(256)
void k_gemm(const bf16* __restrict__ A, const bf16* __restrict__ Bt,
            const float* __restrict__ bias, const float* __restrict__ resid,
            float* __restrict__ Cf, bf16* __restrict__ Cb,
            int M, int N, int K) {
  const int wave = threadIdx.x >> 5;
  const int row0 = blockIdx.y * 128 + (wave >> 1) * 32;
  const int col0 = blockIdx.x * 128 + (wave & 1) * 64;
  if (row0 >= M) return;               // wave-uniform guard (M % 32 == 0)
  const int lane = threadIdx.x & 31;
  v8f acc[2][4];
#pragma unroll
  for (int rt = 0; rt < 2; ++rt)
#pragma unroll
    for (int t = 0; t < 4; ++t)
#pragma unroll
      for (int e = 0; e < 8; ++e) acc[rt][t][e] = 0.f;

  for (int k = 0; k < K; k += 32) {
    v16bf a0 = ld_frag(A, K, row0, k);
    v16bf a1 = ld_frag(A, K, row0 + 16, k);
    if (k + 32 < K) {  // gfx1250 global_prefetch of next A cachelines
      __builtin_prefetch(A + (size_t)(row0 + (lane & 15)) * K + k + 32, 0, 1);
      __builtin_prefetch(A + (size_t)(row0 + 16 + (lane & 15)) * K + k + 32, 0, 1);
    }
#pragma unroll
    for (int t = 0; t < 4; ++t) {
      v16bf b = ld_frag(Bt, K, col0 + t * 16, k);
      acc[0][t] = wmma_bf16(a0, b, acc[0][t]);
      acc[1][t] = wmma_bf16(a1, b, acc[1][t]);
    }
  }
  const int cl = lane & 15;
  const int rh = (lane >> 4) << 3;
#pragma unroll
  for (int rt = 0; rt < 2; ++rt) {
#pragma unroll
    for (int t = 0; t < 4; ++t) {
      const int col = col0 + t * 16 + cl;
      float bi = 0.f;
      if (BIAS) bi = bias[col];
#pragma unroll
      for (int e = 0; e < 8; ++e) {
        const int row = row0 + rt * 16 + rh + e;
        float v = acc[rt][t][e] + bi;
        if (GELU) v = gelu_exact(v);
        if (RESID) v += resid[(size_t)row * N + col];
        if (WF32) Cf[(size_t)row * N + col] = v;
        if (WBF16) Cb[(size_t)row * N + col] = (bf16)v;
      }
    }
  }
}

// ---------------------------------------------------------------- attention
// one block per (window, head): 13 waves, each owns 16 query rows (196 -> 208)
__global__ __launch_bounds__(416)
void k_attn(const bf16* __restrict__ qkv, const float* __restrict__ rph,
            const float* __restrict__ rpw, bf16* __restrict__ aout) {
  const int win = blockIdx.x / 12;
  const int hd  = blockIdx.x % 12;
  const int T0  = win * 196;
  const int w   = threadIdx.x >> 5;    // wave id = row tile 0..12
  const int lane = threadIdx.x & 31;
  const int cl = lane & 15;
  const int rh = (lane >> 4) << 3;

  __shared__ bf16 Ps[13][16][32];      // per-wave P staging (A-layout source)
  __shared__ bf16 Vt[64][224];         // V transposed: [hdim][key], keys padded
  __shared__ bf16 RH[13][16][14];      // rel_h term: q . Rh[qh, kh]
  __shared__ bf16 RW[13][16][14];

  // ---- stage V^T (zero pad keys >= 196)
  for (int i = threadIdx.x; i < 64 * 224; i += 416) {
    int d = i / 224, t = i % 224;
    float v = 0.f;
    if (t < 196) v = (float)qkv[(size_t)(T0 + t) * 2304 + 1536 + hd * 64 + d];
    Vt[d][t] = (bf16)v;
  }
  // ---- per-wave rel-pos tables (unscaled q)
  for (int j = lane; j < 448; j += 32) {
    int tbl = j & 1;
    int q2 = j >> 1;
    int rr = q2 / 14;
    int kk = q2 % 14;
    int mm = w * 16 + rr; if (mm > 195) mm = 195;
    int qh = mm / 14, qw = mm % 14;
    int coord = (tbl ? qw : qh) - kk + 13;
    const float* rp = (tbl ? rpw : rph) + coord * 64;
    const bf16* qr = qkv + (size_t)(T0 + mm) * 2304 + hd * 64;
    float s = 0.f;
    for (int c = 0; c < 64; ++c) s += (float)qr[c] * rp[c];
    if (tbl) RW[w][rr][kk] = (bf16)s; else RH[w][rr][kk] = (bf16)s;
  }
  __syncthreads();

  // ---- scores: q x k^T  (13 col tiles x 2 k-steps)
  v8f acc[13];
#pragma unroll
  for (int ct = 0; ct < 13; ++ct)
#pragma unroll
    for (int e = 0; e < 8; ++e) acc[ct][e] = 0.f;

  v16bf aq0 = ld_frag(qkv, 2304, T0 + w * 16, hd * 64 + 0);
  v16bf aq1 = ld_frag(qkv, 2304, T0 + w * 16, hd * 64 + 32);
#pragma unroll
  for (int ct = 0; ct < 13; ++ct) {
    v16bf b0 = ld_frag(qkv, 2304, T0 + ct * 16, 768 + hd * 64 + 0);
    v16bf b1 = ld_frag(qkv, 2304, T0 + ct * 16, 768 + hd * 64 + 32);
    acc[ct] = wmma_bf16(aq0, b0, acc[ct]);
    acc[ct] = wmma_bf16(aq1, b1, acc[ct]);
  }

  // ---- scale + rel-pos + mask
#pragma unroll
  for (int ct = 0; ct < 13; ++ct) {
    const int col = ct * 16 + cl;
    const bool cv = col < 196;
    const int kh = cv ? col / 14 : 0;
    const int kw = cv ? col % 14 : 0;
#pragma unroll
    for (int e = 0; e < 8; ++e) {
      const int r = rh + e;
      float v = acc[ct][e] * 0.125f + (float)RH[w][r][kh] + (float)RW[w][r][kw];
      acc[ct][e] = cv ? v : -1e30f;
    }
  }

  // ---- softmax over 208 cols; row m lives at (elem e, 16-lane half)
#pragma unroll
  for (int e = 0; e < 8; ++e) {
    float mx = -1e30f;
#pragma unroll
    for (int ct = 0; ct < 13; ++ct) mx = fmaxf(mx, acc[ct][e]);
#pragma unroll
    for (int m = 8; m >= 1; m >>= 1) mx = fmaxf(mx, __shfl_xor(mx, m));
    float sum = 0.f;
#pragma unroll
    for (int ct = 0; ct < 13; ++ct) {
      float p = __expf(acc[ct][e] - mx);
      acc[ct][e] = p;
      sum += p;
    }
#pragma unroll
    for (int m = 8; m >= 1; m >>= 1) sum += __shfl_xor(sum, m);
    float inv = 1.0f / fmaxf(sum, 1e-30f);
#pragma unroll
    for (int ct = 0; ct < 13; ++ct) acc[ct][e] *= inv;
  }

  // ---- out = P x V  (K padded to 224, 7 steps)
  v8f o[4];
#pragma unroll
  for (int t = 0; t < 4; ++t)
#pragma unroll
    for (int e = 0; e < 8; ++e) o[t][e] = 0.f;

#pragma unroll
  for (int s = 0; s < 7; ++s) {
#pragma unroll
    for (int t2 = 0; t2 < 2; ++t2) {
      const int ct = 2 * s + t2;
#pragma unroll
      for (int e = 0; e < 8; ++e) {
        float pv = (ct < 13) ? acc[ct][e] : 0.f;
        Ps[w][rh + e][t2 * 16 + cl] = (bf16)pv;   // D-layout -> LDS
      }
    }
    v16bf a = ld_frag(&Ps[w][0][0], 32, 0, 0);    // same-wave DS in-order
#pragma unroll
    for (int t = 0; t < 4; ++t) {
      v16bf b = ld_frag(&Vt[0][0], 224, t * 16, s * 32);
      o[t] = wmma_bf16(a, b, o[t]);
    }
  }

  // ---- store (only real tokens), head-major columns
#pragma unroll
  for (int t = 0; t < 4; ++t)
#pragma unroll
    for (int e = 0; e < 8; ++e) {
      const int mm = w * 16 + rh + e;
      if (mm < 196)
        aout[(size_t)(T0 + mm) * 768 + hd * 64 + t * 16 + cl] = (bf16)o[t][e];
    }
}

// ---------------------------------------------------------------- unpartition + residual + LN2
__global__ __launch_bounds__(256)
void k_unpart(const float* __restrict__ x, const float* __restrict__ proj,
              const float* __restrict__ g, const float* __restrict__ b,
              float* __restrict__ x1, bf16* __restrict__ mbf) {
  __shared__ float sred[8];
  const int p = blockIdx.x;            // pixel 0..32767
  const int bb = p >> 12, y = (p >> 6) & 63, xx = p & 63;
  const int wy = y / 14, ty = y % 14, wx = xx / 14, tx = xx % 14;
  const int win = bb * 25 + wy * 5 + wx;
  const int tok = ty * 14 + tx;
  const float* pr = proj + ((size_t)win * 196 + tok) * 768;
  const float* xr = x + (size_t)p * 768;
  float* x1r = x1 + (size_t)p * 768;
  bf16* mr = mbf + (size_t)p * 768;
  const int t = threadIdx.x;
  float v0 = xr[t] + pr[t];
  float v1 = xr[t + 256] + pr[t + 256];
  float v2 = xr[t + 512] + pr[t + 512];
  x1r[t] = v0; x1r[t + 256] = v1; x1r[t + 512] = v2;
  float mean = blk_reduce(v0 + v1 + v2, sred, 8) * (1.f / 768.f);
  float d0 = v0 - mean, d1 = v1 - mean, d2 = v2 - mean;
  float var = blk_reduce(d0 * d0 + d1 * d1 + d2 * d2, sred, 8) * (1.f / 768.f);
  float inv = rsqrtf(var + 1e-5f);
  mr[t]       = (bf16)(d0 * inv * g[t]       + b[t]);
  mr[t + 256] = (bf16)(d1 * inv * g[t + 256] + b[t + 256]);
  mr[t + 512] = (bf16)(d2 * inv * g[t + 512] + b[t + 512]);
}

// ---------------------------------------------------------------- conv2 3x3 implicit GEMM
// wave tile 32x64, block 4x2 waves = 128x128 (N=384 -> grid.x=3)
__global__ __launch_bounds__(256)
void k_conv2(const bf16* __restrict__ T1, const bf16* __restrict__ Wt,
             float* __restrict__ Cf) {
  const int wave = threadIdx.x >> 5;
  const int row0 = blockIdx.y * 128 + (wave >> 1) * 32;   // pixel base
  const int col0 = blockIdx.x * 128 + (wave & 1) * 64;
  const int lane = threadIdx.x & 31;
  const int r = lane & 15;
  const int kh8 = (lane >> 4) << 3;
  const int m0 = row0 + r;
  const int m1 = row0 + 16 + r;
  const int pb0 = m0 >> 12, py0 = (m0 >> 6) & 63, px0 = m0 & 63;
  const int pb1 = m1 >> 12, py1 = (m1 >> 6) & 63, px1 = m1 & 63;

  v8f acc[2][4];
#pragma unroll
  for (int rt = 0; rt < 2; ++rt)
#pragma unroll
    for (int t = 0; t < 4; ++t)
#pragma unroll
      for (int e = 0; e < 8; ++e) acc[rt][t][e] = 0.f;

  for (int s = 0; s < 108; ++s) {       // K = 9 taps * 384 ch = 3456
    const int ks = s * 32;
    const int tap = ks / 384;
    const int ci = ks - tap * 384;
    const int dy = tap / 3 - 1, dx = tap % 3 - 1;
    FragU fa0, fa1;
    {
      const int yy = py0 + dy, xx = px0 + dx;
      if ((unsigned)yy < 64u && (unsigned)xx < 64u) {
        const bf16* base =
            T1 + ((size_t)((pb0 << 12) + (yy << 6) + xx)) * 384 + ci + kh8;
        fa0.q[0] = *reinterpret_cast<const uint4*>(base);
        fa0.q[1] = *reinterpret_cast<const uint4*>(base + 16);
      } else {
        fa0.q[0] = make_uint4(0, 0, 0, 0);
        fa0.q[1] = make_uint4(0, 0, 0, 0);
      }
    }
    {
      const int yy = py1 + dy, xx = px1 + dx;
      if ((unsigned)yy < 64u && (unsigned)xx < 64u) {
        const bf16* base =
            T1 + ((size_t)((pb1 << 12) + (yy << 6) + xx)) * 384 + ci + kh8;
        fa1.q[0] = *reinterpret_cast<const uint4*>(base);
        fa1.q[1] = *reinterpret_cast<const uint4*>(base + 16);
      } else {
        fa1.q[0] = make_uint4(0, 0, 0, 0);
        fa1.q[1] = make_uint4(0, 0, 0, 0);
      }
    }
#pragma unroll
    for (int t = 0; t < 4; ++t) {
      v16bf b = ld_frag(Wt, 3456, col0 + t * 16, ks);
      acc[0][t] = wmma_bf16(fa0.v, b, acc[0][t]);
      acc[1][t] = wmma_bf16(fa1.v, b, acc[1][t]);
    }
  }
  const int cl = lane & 15;
  const int rhh = (lane >> 4) << 3;
#pragma unroll
  for (int rt = 0; rt < 2; ++rt)
#pragma unroll
    for (int t = 0; t < 4; ++t)
#pragma unroll
      for (int e = 0; e < 8; ++e)
        Cf[(size_t)(row0 + rt * 16 + rhh + e) * 384 + col0 + t * 16 + cl] =
            acc[rt][t][e];
}

// ---------------------------------------------------------------- channel-first LN (per pixel over C=384) + GELU
__global__ __launch_bounds__(384)
void k_lngelu384(const float* __restrict__ in, const float* __restrict__ g,
                 const float* __restrict__ b, bf16* __restrict__ out) {
  __shared__ float sred[12];
  const float* ir = in + (size_t)blockIdx.x * 384;
  bf16* orow = out + (size_t)blockIdx.x * 384;
  const int t = threadIdx.x;
  float v = ir[t];
  float mean = blk_reduce(v, sred, 12) * (1.f / 384.f);
  float d = v - mean;
  float var = blk_reduce(d * d, sred, 12) * (1.f / 384.f);
  float inv = rsqrtf(var + 1e-6f);
  orow[t] = (bf16)gelu_exact(d * inv * g[t] + b[t]);
}

// ---------------------------------------------------------------- final: LN(conv3 out) + residual
__global__ __launch_bounds__(256)
void k_final(const float* __restrict__ t3, const float* __restrict__ x2,
             const float* __restrict__ g, const float* __restrict__ b,
             float* __restrict__ out) {
  __shared__ float sred[8];
  const size_t base = (size_t)blockIdx.x * 768;
  const int t = threadIdx.x;
  float v0 = t3[base + t], v1 = t3[base + t + 256], v2 = t3[base + t + 512];
  float mean = blk_reduce(v0 + v1 + v2, sred, 8) * (1.f / 768.f);
  float d0 = v0 - mean, d1 = v1 - mean, d2 = v2 - mean;
  float var = blk_reduce(d0 * d0 + d1 * d1 + d2 * d2, sred, 8) * (1.f / 768.f);
  float inv = rsqrtf(var + 1e-6f);
  out[base + t]       = x2[base + t]       + d0 * inv * g[t]       + b[t];
  out[base + t + 256] = x2[base + t + 256] + d1 * inv * g[t + 256] + b[t + 256];
  out[base + t + 512] = x2[base + t + 512] + d2 * inv * g[t + 512] + b[t + 512];
}

// ---------------------------------------------------------------- launcher
extern "C" void kernel_launch(void* const* d_in, const int* in_sizes, int n_in,
                              void* d_out, int out_size, void* d_ws, size_t ws_size,
                              hipStream_t stream) {
  (void)in_sizes; (void)n_in; (void)out_size; (void)ws_size;
  const float* x    = (const float*)d_in[0];
  const float* n1w  = (const float*)d_in[1];
  const float* n1b  = (const float*)d_in[2];
  const float* qkvw = (const float*)d_in[3];
  const float* qkvb = (const float*)d_in[4];
  const float* prjw = (const float*)d_in[5];
  const float* prjb = (const float*)d_in[6];
  const float* rph  = (const float*)d_in[7];
  const float* rpw  = (const float*)d_in[8];
  const float* n2w  = (const float*)d_in[9];
  const float* n2b  = (const float*)d_in[10];
  const float* f1w  = (const float*)d_in[11];
  const float* f1b  = (const float*)d_in[12];
  const float* f2w  = (const float*)d_in[13];
  const float* f2b  = (const float*)d_in[14];
  const float* c1w  = (const float*)d_in[15];
  const float* l1w  = (const float*)d_in[16];
  const float* l1b  = (const float*)d_in[17];
  const float* c2w  = (const float*)d_in[18];
  const float* l2w  = (const float*)d_in[19];
  const float* l2b  = (const float*)d_in[20];
  const float* c3w  = (const float*)d_in[21];
  const float* l3w  = (const float*)d_in[22];
  const float* l3b  = (const float*)d_in[23];

  // ---- workspace arena (with lifetime reuse)
  char* wsp = (char*)d_ws;
  size_t off = 0;
  auto take = [&](size_t bytes) -> char* {
    char* p = wsp + off;
    off += (bytes + 255) & ~(size_t)255;
    return p;
  };
  bf16* qkvWt = (bf16*)take((size_t)2304 * 768 * 2);
  bf16* prjWt = (bf16*)take((size_t)768 * 768 * 2);
  bf16* f1Wt  = (bf16*)take((size_t)3072 * 768 * 2);
  bf16* f2Wt  = (bf16*)take((size_t)768 * 3072 * 2);
  bf16* c1Wt  = (bf16*)take((size_t)384 * 768 * 2);
  bf16* c2Wt  = (bf16*)take((size_t)384 * 3456 * 2);
  bf16* c3Wt  = (bf16*)take((size_t)768 * 384 * 2);
  char* bufA = take((size_t)39200 * 768 * 2);   // h_bf  -> attnout_bf
  char* bufB = take((size_t)32768 * 3072 * 2);  // qkv_bf(+16 pad rows) -> h1_bf
  char* bufC = take((size_t)39200 * 768 * 4);   // projout -> t3
  char* bufD = take((size_t)32768 * 768 * 4);   // x1 -> {t0, t2}
  char* bufE = take((size_t)32768 * 768 * 2);   // m_bf -> x2_bf
  char* bufF = take((size_t)32768 * 768 * 4);   // x2 (live to end)
  char* bufG = take((size_t)32768 * 384 * 2);   // t1_bf
  char* bufH = take((size_t)32768 * 384 * 2);   // t2_bf

  bf16* h_bf    = (bf16*)bufA;
  bf16* aout_bf = (bf16*)bufA;
  bf16* qkv_bf  = (bf16*)bufB;
  bf16* h1_bf   = (bf16*)bufB;
  float* projo  = (float*)bufC;
  float* t3     = (float*)bufC;
  float* x1     = (float*)bufD;
  float* t0     = (float*)bufD;
  float* t2     = (float*)(bufD + (size_t)32768 * 384 * 4);
  bf16* m_bf    = (bf16*)bufE;
  bf16* x2_bf   = (bf16*)bufE;
  float* x2     = (float*)bufF;
  bf16* t1_bf   = (bf16*)bufG;
  bf16* t2_bf   = (bf16*)bufH;

  const dim3 blk(256);
  // ---- weight prep (bf16, N-major)
  k_wt_transpose<<<dim3((768u * 2304u + 255u) / 256u), blk, 0, stream>>>(qkvw, qkvWt, 768, 2304);
  k_wt_transpose<<<dim3((768u * 768u + 255u) / 256u), blk, 0, stream>>>(prjw, prjWt, 768, 768);
  k_wt_transpose<<<dim3((768u * 3072u + 255u) / 256u), blk, 0, stream>>>(f1w, f1Wt, 768, 3072);
  k_wt_transpose<<<dim3((3072u * 768u + 255u) / 256u), blk, 0, stream>>>(f2w, f2Wt, 3072, 768);
  k_wt_convert<<<dim3((384u * 768u + 255u) / 256u), blk, 0, stream>>>(c1w, c1Wt, 384 * 768);
  k_wt_convert<<<dim3((768u * 384u + 255u) / 256u), blk, 0, stream>>>(c3w, c3Wt, 768 * 384);
  k_wt_conv2<<<dim3((384u * 3456u + 255u) / 256u), blk, 0, stream>>>(c2w, c2Wt);

  // ---- 1) LN1 + window partition (39200 padded tokens, bf16)
  k_ln1win<<<dim3(39200), blk, 0, stream>>>(x, n1w, n1b, h_bf);
  // ---- 2) QKV GEMM: [39200 x 2304] = h_bf x qkvW   (bf16 out)
  k_gemm<1, 0, 0, 1, 0><<<dim3(18, 307), blk, 0, stream>>>(
      h_bf, qkvWt, qkvb, nullptr, nullptr, qkv_bf, 39200, 2304, 768);
  // ---- 3) windowed attention with rel-pos (2400 = 200 win x 12 heads)
  k_attn<<<dim3(2400), dim3(416), 0, stream>>>(qkv_bf, rph, rpw, aout_bf);
  // ---- 4) proj GEMM -> f32
  k_gemm<1, 0, 1, 0, 0><<<dim3(6, 307), blk, 0, stream>>>(
      aout_bf, prjWt, prjb, nullptr, projo, nullptr, 39200, 768, 768);
  // ---- 5) unpartition + residual + LN2
  k_unpart<<<dim3(32768), blk, 0, stream>>>(x, projo, n2w, n2b, x1, m_bf);
  // ---- 6) MLP fc1 (+bias, exact GELU) -> bf16
  k_gemm<1, 1, 0, 1, 0><<<dim3(24, 256), blk, 0, stream>>>(
      m_bf, f1Wt, f1b, nullptr, nullptr, h1_bf, 32768, 3072, 768);
  // ---- 7) MLP fc2 (+bias, +residual x1) -> x2 f32 + bf16 mirror
  k_gemm<1, 0, 1, 1, 1><<<dim3(6, 256), blk, 0, stream>>>(
      h1_bf, f2Wt, f2b, x1, x2, x2_bf, 32768, 768, 3072);
  // ---- 8) conv1 1x1 (GEMM, no bias)
  k_gemm<0, 0, 1, 0, 0><<<dim3(3, 256), blk, 0, stream>>>(
      x2_bf, c1Wt, nullptr, nullptr, t0, nullptr, 32768, 384, 768);
  // ---- 9) LN_cf + GELU -> bf16
  k_lngelu384<<<dim3(32768), dim3(384), 0, stream>>>(t0, l1w, l1b, t1_bf);
  // ---- 10) conv2 3x3 implicit GEMM
  k_conv2<<<dim3(3, 256), blk, 0, stream>>>(t1_bf, c2Wt, t2);
  // ---- 11) LN_cf + GELU -> bf16
  k_lngelu384<<<dim3(32768), dim3(384), 0, stream>>>(t2, l2w, l2b, t2_bf);
  // ---- 12) conv3 1x1 (GEMM)
  k_gemm<0, 0, 1, 0, 0><<<dim3(6, 256), blk, 0, stream>>>(
      t2_bf, c3Wt, nullptr, nullptr, t3, nullptr, 32768, 768, 384);
  // ---- 13) final LN_cf + bottleneck residual -> d_out (NHWC f32)
  k_final<<<dim3(32768), blk, 0, stream>>>(t3, x2, l3w, l3b, (float*)d_out);
}
